// FactorizationMachine_19585050869936
// MI455X (gfx1250) — matrix operand order repaired and checked by reading, not compiled, not run on hardware
//
#include <hip/hip_runtime.h>

// CDNA5 / gfx1250, wave32
typedef __attribute__((ext_vector_type(16))) _Float16 v16h;
typedef __attribute__((ext_vector_type(8)))  float    v8f;

#define B_SZ  16384
#define NF_S  26
#define NF_D  13
#define V_SZ  100000
#define D_DIM 64

// ---------------------------------------------------------------------------
// Kernel 1: Vd[B,64] = dense[B,13] @ W[64,13]^T via v_wmma_f32_16x16x32_f16.
// 4 waves per block; wave t computes the 16x16 tile for cols [16t,16t+16).
// K=13 zero-padded to the instruction's K=32 with branchless data masking.
// ---------------------------------------------------------------------------
__global__ __launch_bounds__(128) void fm_dense_proj_wmma(
    const float* __restrict__ dense,   // [B,13]
    const float* __restrict__ W,       // [64,13] row-major (dense_arch_W)
    float* __restrict__ Vd)            // [B,64]
{
    const int lane    = threadIdx.x & 31;
    const int tile    = threadIdx.x >> 5;      // 0..3 -> N-tile
    const int rowBase = blockIdx.x * 16;
    const int m       = lane & 15;
    const bool hi     = lane >= 16;

    // A: 16x32 f16 (M x K). lanes 0-15: M=lane,    halves 0..7 = K 0..7, 8..15 = K 16..23
    //                       lanes 16-31: M=lane-16, halves 0..7 = K 8..15, 8..15 = K 24..31
    // Only K<13 is real. Branchless: clamp address, select-zero the data.
    v16h a;
    const int kbase = hi ? 8 : 0;
    const float* arow = dense + (size_t)(rowBase + m) * NF_D;
    #pragma unroll
    for (int e = 0; e < 8; ++e) {
        const int  k  = kbase + e;
        const bool ok = (k < NF_D);
        const float v = arow[ok ? k : 0];            // always-valid address
        a[e]     = ok ? (_Float16)v : (_Float16)0.0f; // v_cndmask, no exec branch
        a[8 + e] = (_Float16)0.0f;                    // K = 16..31 padded zero
    }

    // B: 32x16 f16 (K x N), B[k][n] = W[n][k].
    // lanes 0-15: N=lane, halves 0..7 = K 0..7; lanes 16-31: halves 0..7 = K 8..15 (only K<13 real); halves 8..15 = K 16..31 -> zero.
    v16h bm = {};
    const int n = tile * 16 + m;
    {
        const float* wrow = W + (size_t)n * NF_D;
        #pragma unroll
        for (int e = 0; e < 8; ++e)
            if (kbase + e < NF_D) bm[e] = (_Float16)wrow[kbase + e];
    }

    v8f c = {};
    c = __builtin_amdgcn_wmma_f32_16x16x32_f16(
            /*neg_a=*/false, a, /*neg_b=*/false, bm,
            /*c_mod=*/(short)0, c, /*reuse_a=*/false, /*reuse_b=*/false);

    // C/D layout: VGPR r -> M=r (lanes 0-15) / M=8+r (lanes 16-31); N = lane&15 (+tile).
    #pragma unroll
    for (int r = 0; r < 8; ++r) {
        const int row = hi ? (8 + r) : r;
        Vd[(size_t)(rowBase + row) * D_DIM + n] = c[r];
    }
}

// ---------------------------------------------------------------------------
// Kernel 2: one wave32 per batch row. Lane owns dims {2*lane, 2*lane+1}
// -> each feature's 256B embedding row is one coalesced float2 load per lane.
// Fully unrolled so all 26 independent gathers can be clause-issued in flight.
// Linear terms folded into the single 5-step wave32 reduction.
// ---------------------------------------------------------------------------
__global__ __launch_bounds__(256) void fm_gather_reduce(
    const int*   __restrict__ sparse_idx, // [B,26]
    const float* __restrict__ dense,      // [B,13]
    const float* __restrict__ emb_lin,    // [26,V]
    const float* __restrict__ emb_v,      // [26,V,64]
    const float* __restrict__ lin_W,      // [1,13]
    const float* __restrict__ lin_b,      // [1]
    const float* __restrict__ arch_b,     // [64]
    const float* __restrict__ Vd,         // [B,64]
    const float* __restrict__ bias,       // [1]
    float*       __restrict__ out)        // [B]
{
    const int lane = threadIdx.x & 31;
    const int wave = threadIdx.x >> 5;
    const int b    = blockIdx.x * 8 + wave;

    // Lanes 0..25: load this row's index for feature `lane`, and gather the
    // first-order sparse term for that feature right away (no per-iter branch).
    int   myIdx   = 0;
    float lin_acc = 0.0f;
    if (lane < NF_S) {
        myIdx   = sparse_idx[(size_t)b * NF_S + lane];
        lin_acc = emb_lin[(size_t)lane * V_SZ + (size_t)myIdx];
    }
    // First-order dense term: lane j<13 contributes dense[b,j]*lin_W[j].
    if (lane < NF_D) lin_acc += dense[(size_t)b * NF_D + lane] * lin_W[lane];

    // v_d (+ dense_arch bias) for this lane's two dims.
    float2 s  = *(const float2*)(Vd + (size_t)b * D_DIM + lane * 2);
    float2 ab = *(const float2*)(arch_b + lane * 2);
    s.x += ab.x; s.y += ab.y;
    float2 q;                               // running sum of squares (incl. v_d)
    q.x = s.x * s.x; q.y = s.y * s.y;

    #pragma unroll
    for (int f = 0; f < NF_S; ++f) {
        const int idx = __shfl(myIdx, f, 32);
        const size_t off = ((size_t)f * V_SZ + (size_t)idx) * D_DIM + lane * 2;
        const float2 e = *(const float2*)(emb_v + off);
        s.x += e.x; s.y += e.y;
        q.x += e.x * e.x; q.y += e.y * e.y;
    }

    // 0.5 * ((sum)^2 - sum of squares) over this lane's dims + linear parts.
    float red = 0.5f * ((s.x * s.x - q.x) + (s.y * s.y - q.y)) + lin_acc;
    #pragma unroll
    for (int sft = 16; sft > 0; sft >>= 1) red += __shfl_xor(red, sft, 32);

    if (lane == 0) out[b] = red + lin_b[0] + bias[0];
}

extern "C" void kernel_launch(void* const* d_in, const int* in_sizes, int n_in,
                              void* d_out, int out_size, void* d_ws, size_t ws_size,
                              hipStream_t stream) {
    const int*   sparse_idx = (const int*)  d_in[0];
    const float* dense      = (const float*)d_in[1];
    const float* emb_lin    = (const float*)d_in[2];
    const float* emb_v      = (const float*)d_in[3];
    const float* lin_W      = (const float*)d_in[4];
    const float* lin_b      = (const float*)d_in[5];
    const float* arch_W     = (const float*)d_in[6];
    const float* arch_b     = (const float*)d_in[7];
    const float* bias       = (const float*)d_in[8];
    float* out = (float*)d_out;
    float* Vd  = (float*)d_ws;                 // B*64*4 = 4 MB scratch, lives in L2

    fm_dense_proj_wmma<<<B_SZ / 16, 128, 0, stream>>>(dense, arch_W, Vd);
    fm_gather_reduce <<<B_SZ / 8, 256, 0, stream>>>(sparse_idx, dense, emb_lin, emb_v,
                                                    lin_W, lin_b, arch_b, Vd, bias, out);
    (void)in_sizes; (void)n_in; (void)out_size; (void)ws_size;
}
